// MambaPatchEmbedding_3753801417223
// MI455X (gfx1250) — compile-verified, matrix operand order checked by reading, not run on hardware
//
#include <hip/hip_runtime.h>
#include <hip/hip_bf16.h>
#include <math.h>

#define ROWS (19*768)   // 14592
#define DM   128
#define DI   256
#define TT   768
#define BB   19

typedef float    v2f  __attribute__((ext_vector_type(2)));
typedef float    v8f  __attribute__((ext_vector_type(8)));
typedef _Float16 v16h __attribute__((ext_vector_type(16)));

// ---------------------------------------------------------------------------
// Generic WMMA GEMM:  C[m,n] = sum_k A[m*lda+k]*W[n*ldw+k] (+bias[n]) (+Cin[m*ldcin+n])
// One wave32 per 16x16 output tile, fp32 V_WMMA_F32_16X16X4_F32, two acc chains.
// Requires K % 8 == 0 (all call sites: 128 or 256). M/N tails masked.
// ---------------------------------------------------------------------------
__global__ __launch_bounds__(256)
void gemm_wmma(const float* __restrict__ A, int lda,
               const float* __restrict__ W, int ldw,
               const float* __restrict__ bias,
               const float* __restrict__ Cin, int ldcin,
               float* __restrict__ C, int ldc,
               int M, int N, int K)
{
  const int wave = threadIdx.x >> 5;
  const int lane = threadIdx.x & 31;
  const int tilesN = (N + 15) >> 4;
  const int tilesM = (M + 15) >> 4;
  const int tile = blockIdx.x * 8 + wave;
  if (tile >= tilesM * tilesN) return;           // wave-uniform exit
  const int tm = tile / tilesN;
  const int tn = tile - tm * tilesN;
  const int m0 = tm << 4, n0 = tn << 4;
  const int r  = lane & 15;
  const int kh = lane >> 4;                       // 0 or 1

  int mA = m0 + r; if (mA > M - 1) mA = M - 1;    // clamp (tail rows recompute row M-1)
  int nB = n0 + r;
  const float bmask = (nB < N) ? 1.0f : 0.0f;     // zero out-of-range weight columns
  if (nB > N - 1) nB = N - 1;
  const float* __restrict__ Arow = A + (size_t)mA * lda;
  const float* __restrict__ Wrow = W + (size_t)nB * ldw;

  v8f acc0 = {}; v8f acc1 = {};
#if __has_builtin(__builtin_amdgcn_wmma_f32_16x16x4_f32)
  // ISA f32 16x4 A layout: lanes0-15 hold K={0,1}, lanes16-31 K={2,3}; B mirrors.
  for (int k = 0; k < K; k += 8) {
    v2f a0, b0, a1, b1;
    a0.x = Arow[k + kh*2];           a0.y = Arow[k + kh*2 + 1];
    b0.x = Wrow[k + kh*2] * bmask;   b0.y = Wrow[k + kh*2 + 1] * bmask;
    a1.x = Arow[k + 4 + kh*2];       a1.y = Arow[k + 4 + kh*2 + 1];
    b1.x = Wrow[k + 4 + kh*2]*bmask; b1.y = Wrow[k + 4 + kh*2 + 1]*bmask;
    acc0 = __builtin_amdgcn_wmma_f32_16x16x4_f32(false, a0, false, b0, (short)0, acc0, false, false);
    acc1 = __builtin_amdgcn_wmma_f32_16x16x4_f32(false, a1, false, b1, (short)0, acc1, false, false);
  }
#elif __has_builtin(__builtin_amdgcn_wmma_f32_16x16x32_f16)
  // fallback: convert to f16, 16x16x32 (A layout: lanes0-15 K=base+{0..7,16..23})
  for (int k = 0; k < K; k += 32) {
    v16h a, b;
    #pragma unroll
    for (int e = 0; e < 8; ++e) {
      a[e]   = (_Float16)Arow[k + kh*8 + e];
      a[e+8] = (_Float16)Arow[k + 16 + kh*8 + e];
      b[e]   = (_Float16)(Wrow[k + kh*8 + e] * bmask);
      b[e+8] = (_Float16)(Wrow[k + 16 + kh*8 + e] * bmask);
    }
    acc0 = __builtin_amdgcn_wmma_f32_16x16x32_f16(false, a, false, b, (short)0, acc0, false, false);
  }
#else
  // scalar fallback (host semantic pass)
  #pragma unroll
  for (int rr = 0; rr < 8; ++rr) {
    int m = m0 + rr + kh*8; if (m > M-1) m = M-1;
    int n = n0 + (lane & 15); if (n > N-1) n = N-1;
    float s = 0.f;
    for (int k = 0; k < K; ++k) s += A[(size_t)m*lda+k]*W[(size_t)n*ldw+k];
    acc0[rr] = s;
  }
#endif
  v8f acc = acc0 + acc1;

  // C/D layout: VGPR rr -> row m0+rr (lanes0-15) / m0+rr+8 (lanes16-31); col = n0+(lane&15)
  const int nS = n0 + (lane & 15);
  if (nS < N) {
    const float bb = bias ? bias[nS] : 0.0f;
    #pragma unroll
    for (int rr = 0; rr < 8; ++rr) {
      const int m = m0 + rr + kh*8;
      if (m < M) {
        float v = acc[rr] + bb;
        if (Cin) v += Cin[(size_t)m*ldcin + nS];
        C[(size_t)m*ldc + nS] = v;
      }
    }
  }
}

// ---------------------------------------------------------------------------
// Front: 1D conv (in-ch=1, k=7, pad 3) + bias + exact GELU, written transposed
// out[(b*768+t)*128 + o]
// ---------------------------------------------------------------------------
__global__ void k_front(const float* __restrict__ x, const float* __restrict__ w1,
                        const float* __restrict__ b1, float* __restrict__ out)
{
  int idx = blockIdx.x * blockDim.x + threadIdx.x;
  if (idx >= ROWS * DM) return;
  int o = idx & 127;
  int row = idx >> 7;
  int t = row % TT, b = row / TT;
  float acc = b1[o];
  #pragma unroll
  for (int k = 0; k < 7; ++k) {
    int tt = t + k - 3;
    if (tt >= 0 && tt < TT) acc += w1[o*7 + k] * x[b*TT + tt];
  }
  out[idx] = 0.5f * acc * (1.0f + erff(acc * 0.70710678118654752f));
}

// layernorm over last dim (128), one wave per row
__global__ void k_layernorm(const float* __restrict__ X, const float* __restrict__ g,
                            const float* __restrict__ bta, float* __restrict__ Y, int rows)
{
  int wave = threadIdx.x >> 5, lane = threadIdx.x & 31;
  int row = blockIdx.x * 8 + wave;
  if (row >= rows) return;
  const float* x = X + (size_t)row * DM;
  float v0 = x[lane], v1 = x[lane+32], v2 = x[lane+64], v3 = x[lane+96];
  float s  = v0+v1+v2+v3;
  float sq = v0*v0+v1*v1+v2*v2+v3*v3;
  #pragma unroll
  for (int m = 16; m; m >>= 1) { s += __shfl_xor(s, m, 32); sq += __shfl_xor(sq, m, 32); }
  float mean = s * (1.0f/128.0f);
  float var  = sq * (1.0f/128.0f) - mean*mean;
  float inv  = rsqrtf(var + 1e-5f);
  float* y = Y + (size_t)row * DM;
  y[lane]    = (v0-mean)*inv*g[lane]    + bta[lane];
  y[lane+32] = (v1-mean)*inv*g[lane+32] + bta[lane+32];
  y[lane+64] = (v2-mean)*inv*g[lane+64] + bta[lane+64];
  y[lane+96] = (v3-mean)*inv*g[lane+96] + bta[lane+96];
}

// causal depthwise conv k=4 over time on x-half of xz, + SiLU  -> xs[row*256+i]
__global__ void k_convsilu(const float* __restrict__ xz, const float* __restrict__ cw,
                           const float* __restrict__ cb, float* __restrict__ xs)
{
  int idx = blockIdx.x * blockDim.x + threadIdx.x;
  if (idx >= ROWS * DI) return;
  int i = idx & 255;
  int row = idx >> 8;
  int t = row % TT, b = row / TT;
  float acc = cb[i];
  #pragma unroll
  for (int k = 0; k < 4; ++k) {
    int tt = t - 3 + k;
    if (tt >= 0) acc += cw[i*4 + k] * xz[((size_t)(b*TT + tt))*512 + i];
  }
  xs[idx] = acc / (1.0f + expf(-acc));   // silu
}

// Selective scan fused with dt-projection, softplus, D skip, SiLU(z) gate.
// One block per batch b (19 blocks, 256 threads = channels). y overwrites x-half of xz.
__global__ __launch_bounds__(256)
void k_scan(const float* __restrict__ dbl, const float* __restrict__ xs,
            float* __restrict__ xz, const float* __restrict__ Alog,
            const float* __restrict__ dtw, const float* __restrict__ dtb,
            const float* __restrict__ Dp)
{
  const int b = blockIdx.x;
  const int d = threadIdx.x;
  __shared__ float sBC[24];
  float Ac[8], w[8], h[8];
  #pragma unroll
  for (int s = 0; s < 8; ++s) { Ac[s] = -expf(Alog[d*8 + s]); w[s] = dtw[d*8 + s]; h[s] = 0.f; }
  const float bt = dtb[d], Dd = Dp[d];
  for (int t = 0; t < TT; ++t) {
    const size_t row = (size_t)b*TT + t;
    if (threadIdx.x < 24) sBC[threadIdx.x] = dbl[row*24 + threadIdx.x];
    __syncthreads();
    float dl = bt;
    #pragma unroll
    for (int s = 0; s < 8; ++s) dl += sBC[s] * w[s];
    float dt = (dl > 20.0f) ? dl : log1pf(expf(dl));          // softplus
    float xv = xs[row*DI + d];
    float dtx = dt * xv;
    float y = 0.f;
    #pragma unroll
    for (int s = 0; s < 8; ++s) {
      h[s] = expf(dt * Ac[s]) * h[s] + dtx * sBC[8 + s];      // h = dA*h + dB*x
      y += h[s] * sBC[16 + s];                                // y = <h, C>
    }
    y += Dd * xv;
    float z = xz[row*512 + DI + d];
    y *= z / (1.0f + expf(-z));                               // * silu(z)
    __syncthreads();
    xz[row*512 + d] = y;                                      // reuse dead x-slot
  }
}

// time flip per batch: out[b,t,c] = in[b,767-t,c]
__global__ void k_flip(const float* __restrict__ in, float* __restrict__ out)
{
  int idx = blockIdx.x * blockDim.x + threadIdx.x;
  if (idx >= ROWS * DM) return;
  int c = idx & 127;
  int row = idx >> 7;
  int t = row % TT, b = row / TT;
  out[idx] = in[((size_t)b*TT + (TT-1-t))*DM + c];
}

// gather band tokens: rows [0,114)=band0, [114,152)=band1, [152,171)=band2
__global__ void k_gather(const float* __restrict__ fused, float* __restrict__ tok)
{
  int idx = blockIdx.x * blockDim.x + threadIdx.x;
  if (idx >= 171 * DM) return;
  int c = idx & 127;
  int row = idx >> 7;
  int b, t;
  if (row < 114)      { b = row / 6;          t = 128*(row % 6 + 1) - 1; }
  else if (row < 152) { int rr = row - 114; b = rr / 2; t = 384*(rr % 2 + 1) - 1; }
  else                { b = row - 152;        t = TT - 1; }
  tok[idx] = fused[((size_t)b*TT + t)*DM + c];
}

__global__ void k_cbias(const float* __restrict__ bp_b, const float* __restrict__ bemb,
                        float* __restrict__ cb)
{
  int idx = blockIdx.x * blockDim.x + threadIdx.x;
  if (idx < 3*DM) cb[idx] = bp_b[idx] + bemb[idx];
}

// scatter projected tokens to interleaved pe (Nt=9): band0 j->{0,1,2,4,5,6}, band1->{3,7}, band2->8
__global__ void k_scatter(const float* __restrict__ tokout, float* __restrict__ pe)
{
  int idx = blockIdx.x * blockDim.x + threadIdx.x;
  if (idx >= 171 * DM) return;
  int c = idx & 127;
  int row = idx >> 7;
  int b, pos;
  if (row < 114)      { b = row / 6; const int map0[6] = {0,1,2,4,5,6}; pos = map0[row % 6]; }
  else if (row < 152) { int rr = row - 114; b = rr / 2; pos = (rr % 2) ? 7 : 3; }
  else                { b = row - 152; pos = 8; }
  pe[((size_t)b*9 + pos)*DM + c] = tokout[idx];
}

// depthwise 19x7 pos conv (pad 9,3) over (C=19, Nt=9) + bias + residual -> d_out
__global__ void k_posconv(const float* __restrict__ pe, const float* __restrict__ pw,
                          const float* __restrict__ pb, float* __restrict__ out)
{
  int idx = blockIdx.x * blockDim.x + threadIdx.x;
  if (idx >= 19*9*DM) return;
  int ch = idx & 127;
  int ij = idx >> 7;
  int j = ij % 9, i = ij / 9;
  float acc = pb[ch];
  for (int a = 0; a < 19; ++a) {
    int y = i + a - 9;
    if (y < 0 || y >= 19) continue;
    #pragma unroll
    for (int c = 0; c < 7; ++c) {
      int xj = j + c - 3;
      if (xj < 0 || xj >= 9) continue;
      acc += pw[ch*133 + a*7 + c] * pe[((size_t)y*9 + xj)*DM + ch];
    }
  }
  out[idx] = pe[(size_t)ij*DM + ch] + acc;
}

// ---------------------------------------------------------------------------
static inline void launch_gemm(const float* A, int lda, const float* W, int ldw,
                               const float* bias, const float* Cin, int ldcin,
                               float* C, int ldc, int M, int N, int K, hipStream_t s)
{
  int tiles = ((M + 15)/16) * ((N + 15)/16);
  int blocks = (tiles + 7)/8;
  gemm_wmma<<<blocks, 256, 0, s>>>(A, lda, W, ldw, bias, Cin, ldcin, C, ldc, M, N, K);
}

extern "C" void kernel_launch(void* const* d_in, const int* in_sizes, int n_in,
                              void* d_out, int out_size, void* d_ws, size_t ws_size,
                              hipStream_t stream)
{
  const float* x      = (const float*)d_in[0];
  const float* sp_w1  = (const float*)d_in[1];
  const float* sp_b1  = (const float*)d_in[2];
  const float* sp_w2  = (const float*)d_in[3];
  const float* sp_b2  = (const float*)d_in[4];
  // per-direction param blocks: f_* at 5..15, b_* at 16..26
  const float* fuse_w  = (const float*)d_in[27];
  const float* fuse_b  = (const float*)d_in[28];
  const float* band_emb= (const float*)d_in[29];
  const float* bp_w    = (const float*)d_in[30];
  const float* bp_b    = (const float*)d_in[31];
  const float* pos_w   = (const float*)d_in[32];
  const float* pos_b   = (const float*)d_in[33];

  float* ws = (float*)d_ws;
  size_t o = 0;
  float* FEATS = ws + o; o += (size_t)ROWS*DM;     // 1.87M
  float* HF    = ws + o; o += (size_t)ROWS*DM;
  float* HB    = ws + o; o += (size_t)ROWS*DM;
  float* LNB   = ws + o; o += (size_t)ROWS*DM;     // ln buf / hgT / hb-flip
  float* XZ    = ws + o; o += (size_t)ROWS*512;
  float* XS    = ws + o; o += (size_t)ROWS*DI;
  float* DBL   = ws + o; o += (size_t)ROWS*24;
  float* FUSED = ws + o; o += (size_t)ROWS*DM;
  float* TOK   = ws + o; o += 171*DM;
  float* TOKO  = ws + o; o += 171*DM;
  float* PE    = ws + o; o += 171*DM;              // 19*9*128
  float* CB    = ws + o; o += 3*DM;

  const int EW = 256;
  // 1) front conv + gelu (transposed) -> LNB ; 2) channel mix -> FEATS
  k_front<<<(ROWS*DM + EW-1)/EW, EW, 0, stream>>>(x, sp_w1, sp_b1, LNB);
  launch_gemm(LNB, DM, sp_w2, DM, sp_b2, nullptr, 0, FEATS, DM, ROWS, DM, DM, stream);

  hipMemcpyAsync(HF, FEATS, (size_t)ROWS*DM*sizeof(float), hipMemcpyDeviceToDevice, stream);
  k_flip<<<(ROWS*DM + EW-1)/EW, EW, 0, stream>>>(FEATS, HB);

  for (int dir = 0; dir < 2; ++dir) {
    const int base = 5 + dir*11;
    const float* ln_g  = (const float*)d_in[base+0];
    const float* ln_b  = (const float*)d_in[base+1];
    const float* in_w  = (const float*)d_in[base+2];
    const float* cw    = (const float*)d_in[base+3];
    const float* cbp   = (const float*)d_in[base+4];
    const float* xp_w  = (const float*)d_in[base+5];
    const float* dt_w  = (const float*)d_in[base+6];
    const float* dt_b  = (const float*)d_in[base+7];
    const float* Alog  = (const float*)d_in[base+8];
    const float* Dp    = (const float*)d_in[base+9];
    const float* out_w = (const float*)d_in[base+10];
    float* H = dir ? HB : HF;
    for (int i = 0; i < 2; ++i) {
      k_layernorm<<<(ROWS + 7)/8, 256, 0, stream>>>(H, ln_g + i*DM, ln_b + i*DM, LNB, ROWS);
      launch_gemm(LNB, DM, in_w + (size_t)i*512*DM, DM, nullptr, nullptr, 0,
                  XZ, 512, ROWS, 512, DM, stream);
      k_convsilu<<<(ROWS*DI + EW-1)/EW, EW, 0, stream>>>(XZ, cw + i*DI*4, cbp + i*DI, XS);
      launch_gemm(XS, DI, xp_w + (size_t)i*24*DI, DI, nullptr, nullptr, 0,
                  DBL, 24, ROWS, 24, DI, stream);
      k_scan<<<BB, 256, 0, stream>>>(DBL, XS, XZ, Alog + i*DI*8, dt_w + i*DI*8,
                                     dt_b + i*DI, Dp + i*DI);
      launch_gemm(XZ, 512, out_w + (size_t)i*DM*DI, DI, nullptr, H, DM,
                  H, DM, ROWS, DM, DI, stream);   // residual add
    }
  }

  // fuse: concat(hf, flip(hb)) x fuse_w^T + fuse_b, done as 2 accumulating passes
  k_flip<<<(ROWS*DM + EW-1)/EW, EW, 0, stream>>>(HB, LNB);
  launch_gemm(HF,  DM, fuse_w,        2*DM, fuse_b,  nullptr, 0, FUSED, DM, ROWS, DM, DM, stream);
  launch_gemm(LNB, DM, fuse_w + DM,   2*DM, nullptr, FUSED, DM, FUSED, DM, ROWS, DM, DM, stream);

  // band pooling
  k_gather<<<(171*DM + EW-1)/EW, EW, 0, stream>>>(FUSED, TOK);
  k_cbias<<<2, 256, 0, stream>>>(bp_b, band_emb, CB);
  launch_gemm(TOK,          DM, bp_w,           DM, CB,        nullptr, 0, TOKO,          DM, 114, DM, DM, stream);
  launch_gemm(TOK + 114*DM, DM, bp_w + DM*DM,   DM, CB + DM,   nullptr, 0, TOKO + 114*DM, DM,  38, DM, DM, stream);
  launch_gemm(TOK + 152*DM, DM, bp_w + 2*DM*DM, DM, CB + 2*DM, nullptr, 0, TOKO + 152*DM, DM,  19, DM, DM, stream);
  k_scatter<<<(171*DM + EW-1)/EW, EW, 0, stream>>>(TOKO, PE);

  // pos conv + residual -> output
  k_posconv<<<(19*9*DM + EW-1)/EW, EW, 0, stream>>>(PE, pos_w, pos_b, (float*)d_out);
}
